// VisionMambaBlock_46995532152933
// MI455X (gfx1250) — matrix operand order repaired and checked by reading, not compiled, not run on hardware
//
#include <hip/hip_runtime.h>

// ---------------------------------------------------------------------------
// VisionMambaBlock on MI455X (gfx1250): bf16 WMMA GEMM pipeline + SSM scan.
// Matrix math: v_wmma_f32_16x16x32_bf16.
//   - weight GEMMs: TDM tensor_load_to_lds staging (TENSORcnt), LDS dbl-buffer
//   - final activation GEMM: async global->LDS copies (ASYNCcnt) for A tile
// ---------------------------------------------------------------------------

constexpr int kB  = 4;
constexpr int kS  = 1024;
constexpr int kD  = 1024;
constexpr int kDT = 64;     // DT_RANK
constexpr int kN  = 16;     // state dim N
constexpr int kR  = 96;     // DT_RANK + 2N
constexpr int kM  = kB * kS;  // 4096 rows for the token-parallel GEMMs

typedef float  f32x8   __attribute__((ext_vector_type(8)));
typedef __bf16 bf16x16 __attribute__((ext_vector_type(16)));
typedef int    i32x4   __attribute__((ext_vector_type(4)));
typedef int    i32x8   __attribute__((ext_vector_type(8)));

union FragB16 { bf16x16 v; i32x4 q[2]; };

__device__ __forceinline__ float softplus_f(float t) {
  return (t > 20.0f) ? t : log1pf(__expf(t));
}

// --------------------- CDNA5 async global->LDS (ASYNCcnt) ------------------
__device__ __forceinline__ void async_copy_b128(uint32_t lds_byte_addr,
                                                const void* gptr) {
  const uint64_t ga = (uint64_t)(uintptr_t)gptr;
  asm volatile("global_load_async_to_lds_b128 %0, %1, off"
               :: "v"(lds_byte_addr), "v"(ga) : "memory");
}
__device__ __forceinline__ void async_copy_b128_off16(uint32_t lds_byte_addr,
                                                      const void* gptr) {
  const uint64_t ga = (uint64_t)(uintptr_t)gptr;
  asm volatile("global_load_async_to_lds_b128 %0, %1, off offset:16"
               :: "v"(lds_byte_addr), "v"(ga) : "memory");
}
__device__ __forceinline__ void wait_async0() {
  asm volatile("s_wait_asynccnt 0x0" ::: "memory");
}

// --------------------- CDNA5 Tensor Data Mover (TENSORcnt) -----------------
// 2D tile DMA: global (strided rows) -> LDS (packed tile_x * tile_y).
// D# packing per ISA 08 §8.3/8.4:
//  g0: [1:0]=count=1 | lds_addr | global_addr[56:0] | type=2
//  g1: data_size=1 (2B), tensor_dim0/1, tile_dim0/1 (tile_dim2=0 -> 2D),
//      tensor_dim0_stride (48b). Rows >= tensor_dim1 read as zero (OOB fill).
__device__ __forceinline__ void tdm_load_2d(uint32_t lds_byte, const void* gptr,
                                            uint32_t tile_x, uint32_t tile_y,
                                            uint32_t tens_x, uint32_t tens_y,
                                            uint32_t stride_elems) {
  const uint64_t ga = (uint64_t)(uintptr_t)gptr;
  i32x4 g0;
  g0[0] = 1;                                            // count=1 (valid user D#)
  g0[1] = (int)lds_byte;                                // lds_addr (bytes)
  g0[2] = (int)(uint32_t)ga;                            // global_addr[31:0]
  g0[3] = (int)((uint32_t)(ga >> 32) | (2u << 30));     // addr[56:32] | type=2
  i32x8 g1;
  g1[0] = 0x00010000;                                   // data_size=2B, mask=0
  g1[1] = (int)(tens_x << 16);                          // tensor_dim0[15:0]
  g1[2] = (int)((tens_x >> 16) | (tens_y << 16));       // dim0 hi | dim1 lo
  g1[3] = (int)((tens_y >> 16) | (tile_x << 16));       // dim1 hi | tile_dim0
  g1[4] = (int)tile_y;                                  // tile_dim1, tile_dim2=0
  g1[5] = (int)stride_elems;                            // dim0_stride[31:0]
  g1[6] = 0;                                            // stride hi, dim1_stride lo
  g1[7] = 0;
  asm volatile("tensor_load_to_lds %0, %1" :: "s"(g0), "s"(g1) : "memory");
}

// ---------------------------------------------------------------------------
// Fused LayerNorm + SiLU: one block per token row of D=1024.
// Emits xn (bf16, GEMM A operand) and z = silu(xn) (bf16, final GEMM B operand)
// ---------------------------------------------------------------------------
__global__ __launch_bounds__(256)
void ln_silu_kernel(const float* __restrict__ x, const float* __restrict__ w,
                    const float* __restrict__ b, __bf16* __restrict__ xn,
                    __bf16* __restrict__ z)
{
  const int row = blockIdx.x;
  const int tid = threadIdx.x;
  const float* xr = x + (size_t)row * kD;
  float vals[4];
  float s = 0.f, sq = 0.f;
#pragma unroll
  for (int i = 0; i < 4; ++i) {
    const float v = xr[tid + i * 256];
    vals[i] = v; s += v; sq += v * v;
  }
  __shared__ float r1[256], r2[256];
  r1[tid] = s; r2[tid] = sq;
  __syncthreads();
  for (int st = 128; st > 0; st >>= 1) {
    if (tid < st) { r1[tid] += r1[tid + st]; r2[tid] += r2[tid + st]; }
    __syncthreads();
  }
  const float mu  = r1[0] * (1.0f / kD);
  const float var = r2[0] * (1.0f / kD) - mu * mu;
  const float rs  = rsqrtf(var + 1e-5f);
#pragma unroll
  for (int i = 0; i < 4; ++i) {
    const int c = tid + i * 256;
    const float nv = (vals[i] - mu) * rs * w[c] + b[c];
    const size_t idx = (size_t)row * kD + c;
    xn[idx] = (__bf16)nv;
    z[idx]  = (__bf16)(nv / (1.0f + __expf(-nv)));   // SiLU
  }
}

// ---------------------------------------------------------------------------
__global__ void cvt_f32_bf16(const float* __restrict__ in,
                             __bf16* __restrict__ out, int n)
{
  const int i = blockIdx.x * 256 + threadIdx.x;
  if (i < n) out[i] = (__bf16)in[i];
}

// ---------------------------------------------------------------------------
// bf16 WMMA GEMM:  C[M,N] = A[M,K] @ B^T  (+bias, +softplus, +skip)
//   BT=false : B is a weight matrix W[N,K]. TDM tensor_load_to_lds stages both
//              tiles (issued by wave 0 only; TDM ignores EXEC but issues per
//              wave); LDS double-buffered, next DMA overlaps current WMMAs.
//   BT=true  : B is an activation Z[K,N]; A tile staged with async global->LDS
//              copies, Z transposed into LDS synchronously.
// Workgroup: 256 thr = 8 waves, tile 128(M) x 64(N) x 32(K); each wave owns
// a 32x32 sub-tile = 2x2 WMMA accumulators.
// Fragment layout per CDNA5 ISA 7.12.2 (16-bit A: two 8-elem K chunks/lane;
// B: same two 8-elem K chunks/lane as A, read from the N-major B tile).
// flags: bit0 = add bias[col], bit1 = softplus. skip!=null => += skip[idx].
// ---------------------------------------------------------------------------
template <bool BT>
__global__ __launch_bounds__(256)
void gemm_bf16_wmma(const __bf16* __restrict__ A, long lda, long sA,
                    const __bf16* __restrict__ Bmat, long ldb, long sB,
                    float* __restrict__ Cf, __bf16* __restrict__ Cb,
                    long ldc, long sC,
                    const float* __restrict__ bias,
                    const float* __restrict__ skip, long sS,
                    int M, int N, int K, int flags)
{
  constexpr int BM = 128, BN = 64, BK = 32;
  constexpr int BP   = BT ? 40 : 32;  // LDS pitch for B tile (pad when transposing)
  constexpr int NBUF = BT ? 1 : 2;    // double buffer for the TDM path
  __shared__ __bf16 As[NBUF * BM * BK];
  __shared__ __bf16 Bs[NBUF * BN * BP];

  const int tid  = threadIdx.x;
  const int lane = tid & 31;
  const int wv   = tid >> 5;
  const int wm   = wv & 3;           // 4 wave rows of 32
  const int wn   = wv >> 2;          // 2 wave cols of 32
  const int lh   = lane >> 4;        // half-wave: K-chunk selector
  const int ll   = lane & 15;        // row (A) / col (B) within 16

  const int bz = blockIdx.z;
  const __bf16* Ab = A    + (size_t)bz * sA;
  const __bf16* Bb = Bmat + (size_t)bz * sB;
  float*  Cfp = Cf ? Cf + (size_t)bz * sC : nullptr;
  __bf16* Cbp = Cb ? Cb + (size_t)bz * sC : nullptr;
  const float* Sk = skip ? skip + (size_t)bz * sS : nullptr;

  const int row0 = blockIdx.y * BM;
  const int col0 = blockIdx.x * BN;

  f32x8 acc[2][2];
#pragma unroll
  for (int i = 0; i < 2; ++i)
#pragma unroll
    for (int j = 0; j < 2; ++j)
#pragma unroll
      for (int e = 0; e < 8; ++e) acc[i][j][e] = 0.f;

  // cooperative-copy work assignments (16B aligned by construction)
  const int ar = tid >> 1, ac = (tid & 1) * 16;   // A: 128 rows x 32 cols
  const int zk = tid >> 3, zc = (tid & 7) * 8;    // Z: 32 K x 64 N

  const uint32_t as_lds = (uint32_t)(uintptr_t)(void*)As;
  const uint32_t bs_lds = (uint32_t)(uintptr_t)(void*)Bs;

  if constexpr (!BT) {
    // ------------------- TDM double-buffered weight path -------------------
    const uint32_t tensNB = (uint32_t)((N - col0 < BN) ? (N - col0) : BN);

    auto stage = [&](int k0s, int pb) {   // one DMA per tile, wave 0 only
      tdm_load_2d(as_lds + (uint32_t)(pb * BM * BK) * 2u,
                  Ab + (size_t)row0 * lda + k0s,
                  BK, BM, BK, BM, (uint32_t)lda);
      tdm_load_2d(bs_lds + (uint32_t)(pb * BN * BP) * 2u,
                  Bb + (size_t)col0 * ldb + k0s,
                  BK, BN, BK, tensNB, (uint32_t)ldb);  // OOB rows -> zero fill
    };

    if (wv == 0) {
      stage(0, 0);
      __builtin_amdgcn_s_wait_tensorcnt(0);
    }
    __syncthreads();

    int pb = 0;
    for (int k0 = 0; k0 < K; k0 += BK) {
      const bool more = (k0 + BK) < K;
      if (more && wv == 0) stage(k0 + BK, pb ^ 1);  // overlap DMA with WMMAs

      const __bf16* Asb = As + pb * BM * BK;
      const __bf16* Bsb = Bs + pb * BN * BP;
      FragB16 af[2], bf2[2];
#pragma unroll
      for (int i = 0; i < 2; ++i) {
        const __bf16* p = Asb + (wm * 32 + i * 16 + ll) * BK;
        af[i].q[0] = *(const i32x4*)(p + lh * 8);
        af[i].q[1] = *(const i32x4*)(p + 16 + lh * 8);
      }
#pragma unroll
      for (int j = 0; j < 2; ++j) {
        const __bf16* p = Bsb + (wn * 32 + j * 16 + ll) * BP;
        bf2[j].q[0] = *(const i32x4*)(p + lh * 8);
        bf2[j].q[1] = *(const i32x4*)(p + 16 + lh * 8);
      }
#pragma unroll
      for (int i = 0; i < 2; ++i)
#pragma unroll
        for (int j = 0; j < 2; ++j)
          acc[i][j] = __builtin_amdgcn_wmma_f32_16x16x32_bf16(
              false, af[i].v, false, bf2[j].v, (short)0, acc[i][j], false, false);

      if (more && wv == 0) __builtin_amdgcn_s_wait_tensorcnt(0);
      __syncthreads();           // compiler adds s_wait_dscnt; fences buffers
      pb ^= 1;
    }
  } else {
    // -------- async A staging + synchronous transposing Z staging ----------
    for (int k0 = 0; k0 < K; k0 += BK) {
      {  // A tile via CDNA5 async copy (contiguous 32B per lane)
        const __bf16* asrc = Ab + (size_t)(row0 + ar) * lda + k0 + ac;
        const uint32_t adst = as_lds + (uint32_t)(ar * BK + ac) * 2u;
        async_copy_b128(adst, asrc);
        async_copy_b128_off16(adst, asrc);
      }
      {  // Z tile transpose (2-byte scatter: neither async nor TDM can)
        const i32x4 v = *(const i32x4*)(Bb + (size_t)(k0 + zk) * ldb + col0 + zc);
        const __bf16* pv = (const __bf16*)&v;
#pragma unroll
        for (int j = 0; j < 8; ++j) Bs[(zc + j) * BP + zk] = pv[j];
      }
      if (k0 + BK < K)
        __builtin_prefetch(Bb + (size_t)(k0 + BK + zk) * ldb + col0 + zc, 0, 3);
      wait_async0();
      __syncthreads();

      FragB16 af[2], bf2[2];
#pragma unroll
      for (int i = 0; i < 2; ++i) {
        const __bf16* p = As + (wm * 32 + i * 16 + ll) * BK;
        af[i].q[0] = *(const i32x4*)(p + lh * 8);
        af[i].q[1] = *(const i32x4*)(p + 16 + lh * 8);
      }
#pragma unroll
      for (int j = 0; j < 2; ++j) {
        const __bf16* p = Bs + (wn * 32 + j * 16 + ll) * BP;
        bf2[j].q[0] = *(const i32x4*)(p + lh * 8);
        bf2[j].q[1] = *(const i32x4*)(p + 16 + lh * 8);
      }
#pragma unroll
      for (int i = 0; i < 2; ++i)
#pragma unroll
        for (int j = 0; j < 2; ++j)
          acc[i][j] = __builtin_amdgcn_wmma_f32_16x16x32_bf16(
              false, af[i].v, false, bf2[j].v, (short)0, acc[i][j], false, false);
      __syncthreads();
    }
  }

  // Epilogue. C 16x16 f32 layout: VGPR v -> row v (lanes0-15) / v+8 (16-31)
#pragma unroll
  for (int i = 0; i < 2; ++i) {
#pragma unroll
    for (int j = 0; j < 2; ++j) {
      const int cc = col0 + wn * 32 + j * 16 + ll;
      if (cc >= N) continue;
      const int rb = row0 + wm * 32 + i * 16 + lh * 8;
#pragma unroll
      for (int v = 0; v < 8; ++v) {
        float t = acc[i][j][v];
        if (flags & 1) t += bias[cc];
        if (flags & 2) t = softplus_f(t);
        const size_t idx = (size_t)(rb + v) * ldc + cc;
        if (Sk)  t += Sk[idx];
        if (Cfp) Cfp[idx] = t;
        if (Cbp) Cbp[idx] = (__bf16)t;
      }
    }
  }
}

// ---------------------------------------------------------------------------
// Selective scan: one lane per (b, d, n); 16 lanes (n=0..15) form a channel
// group, reduced each step with __shfl_xor (wave32: masks 1/2/4/8 stay in the
// 16-lane half).  h_l = exp(delta*A)*h_{l-1} + delta*x*B;  y = sum_n h*C + D*x
// Second branch accumulates y1+y2 and emits bf16 for the final WMMA GEMM.
// ---------------------------------------------------------------------------
__global__ __launch_bounds__(256)
void ssm_scan_kernel(const float* __restrict__ delta, const float* __restrict__ xc,
                     const float* __restrict__ dbc,  const float* __restrict__ A_log,
                     const float* __restrict__ Dp,
                     const float* __restrict__ y_prev,
                     float* __restrict__ y_f32, __bf16* __restrict__ y_bf16)
{
  const int gid = blockIdx.x * 256 + threadIdx.x;
  const int n   = gid & 15;
  const int ch  = gid >> 4;          // 0 .. B*D-1
  const int d   = ch % kD;
  const int bb  = ch / kD;
  const float Aval = -__expf(A_log[d * kN + n]);
  const float dpv  = Dp[d];
  float h = 0.f;
  for (int l = 0; l < kS; ++l) {
    const size_t tix = (size_t)bb * kS + l;
    const size_t idx = tix * kD + d;
    const float dl = delta[idx];             // broadcast across 16 lanes
    const float xv = xc[idx];
    const float Bv = dbc[tix * kR + kDT + n];
    const float Cv = dbc[tix * kR + kDT + kN + n];
    const float dA = __expf(dl * Aval);
    h = dA * h + dl * xv * Bv;
    float p = h * Cv;
    p += __shfl_xor(p, 1, 32);
    p += __shfl_xor(p, 2, 32);
    p += __shfl_xor(p, 4, 32);
    p += __shfl_xor(p, 8, 32);
    if (n == 0) {
      float yv = p + dpv * xv;
      if (y_prev) yv += y_prev[idx];
      if (y_f32)  y_f32[idx]  = yv;
      if (y_bf16) y_bf16[idx] = (__bf16)yv;
    }
  }
}

// ---------------------------------------------------------------------------
extern "C" void kernel_launch(void* const* d_in, const int* in_sizes, int n_in,
                              void* d_out, int out_size, void* d_ws, size_t ws_size,
                              hipStream_t stream)
{
  (void)in_sizes; (void)n_in; (void)out_size; (void)ws_size;
  const float* x    = (const float*)d_in[0];
  const float* Wf   = (const float*)d_in[1];
  const float* bf_  = (const float*)d_in[2];
  const float* Wb   = (const float*)d_in[3];
  const float* bb_  = (const float*)d_in[4];
  const float* lnw  = (const float*)d_in[5];
  const float* lnb  = (const float*)d_in[6];
  const float* Wdbc = (const float*)d_in[7];
  const float* Wdt  = (const float*)d_in[8];
  const float* bdt  = (const float*)d_in[9];
  const float* Alog = (const float*)d_in[10];
  const float* Dp   = (const float*)d_in[11];
  float* out = (float*)d_out;

  // workspace carve-up (~91 MB), 256B aligned
  char* wsp = (char*)d_ws;
  auto alloc = [&](size_t bytes) {
    char* p = wsp;
    wsp += (bytes + 255) & ~(size_t)255;
    return p;
  };
  __bf16* xn_b   = (__bf16*)alloc((size_t)kM * kD * 2);
  __bf16* z_b    = (__bf16*)alloc((size_t)kM * kD * 2);
  __bf16* wf_b   = (__bf16*)alloc((size_t)kD * kD * 2);
  __bf16* wb_b   = (__bf16*)alloc((size_t)kD * kD * 2);
  __bf16* wdbc_b = (__bf16*)alloc((size_t)kR * kD * 2);
  __bf16* wdt_b  = (__bf16*)alloc((size_t)kD * kDT * 2);
  float*  xc_f   = (float*) alloc((size_t)kM * kD * 4);   // reused by both branches
  __bf16* xc_b   = (__bf16*)alloc((size_t)kM * kD * 2);
  float*  dbc_f  = (float*) alloc((size_t)kM * kR * 4);
  __bf16* dbc_b  = (__bf16*)alloc((size_t)kM * kR * 2);
  float*  del_f  = (float*) alloc((size_t)kM * kD * 4);
  float*  y1_f   = (float*) alloc((size_t)kM * kD * 4);
  __bf16* ys_b   = (__bf16*)alloc((size_t)kM * kD * 2);   // y1+y2, bf16

  // 1) LayerNorm + SiLU
  ln_silu_kernel<<<kM, 256, 0, stream>>>(x, lnw, lnb, xn_b, z_b);

  // 2) weight conversions to bf16
  cvt_f32_bf16<<<(kD * kD + 255) / 256, 256, 0, stream>>>(Wf, wf_b, kD * kD);
  cvt_f32_bf16<<<(kD * kD + 255) / 256, 256, 0, stream>>>(Wb, wb_b, kD * kD);
  cvt_f32_bf16<<<(kR * kD + 255) / 256, 256, 0, stream>>>(Wdbc, wdbc_b, kR * kD);
  cvt_f32_bf16<<<(kD * kDT + 255) / 256, 256, 0, stream>>>(Wdt, wdt_b, kD * kDT);

  // 3) two SSM branches (forward conv / backward conv), shared SSM weights
  for (int brn = 0; brn < 2; ++brn) {
    const __bf16* wmat = brn ? wb_b : wf_b;
    const float*  bias = brn ? bb_  : bf_;

    // x_c = xn @ W.T + b     (4096 x 1024 x 1024), also chained bf16 copy
    dim3 g1((kD + 63) / 64, (kM + 127) / 128, 1);
    gemm_bf16_wmma<false><<<g1, 256, 0, stream>>>(
        xn_b, kD, 0, wmat, kD, 0, xc_f, xc_b, kD, 0, bias, nullptr, 0,
        kM, kD, kD, /*flags=*/1);

    // dbc = x_c @ W_dbc.T    (4096 x 96 x 1024)
    dim3 g2((kR + 63) / 64, (kM + 127) / 128, 1);
    gemm_bf16_wmma<false><<<g2, 256, 0, stream>>>(
        xc_b, kD, 0, wdbc_b, kD, 0, dbc_f, dbc_b, kR, 0, nullptr, nullptr, 0,
        kM, kR, kD, /*flags=*/0);

    // delta = softplus(dbc[:, :64] @ W_dt.T + b_dt)  (4096 x 1024 x 64)
    dim3 g3((kD + 63) / 64, (kM + 127) / 128, 1);
    gemm_bf16_wmma<false><<<g3, 256, 0, stream>>>(
        dbc_b, kR, 0, wdt_b, kDT, 0, del_f, nullptr, kD, 0, bdt, nullptr, 0,
        kM, kD, kDT, /*flags=*/3);

    // selective scan; branch 1 accumulates y1+y2 -> bf16
    ssm_scan_kernel<<<(kB * kD * 16) / 256, 256, 0, stream>>>(
        del_f, xc_f, dbc_f, Alog, Dp,
        brn ? y1_f : nullptr,
        brn ? nullptr : y1_f,
        brn ? ys_b : nullptr);
  }

  // 4) out = (y1+y2) @ z + x   (batched, b=4, 1024x1024x1024 each)
  dim3 gf((kD + 63) / 64, (kS + 127) / 128, kB);
  gemm_bf16_wmma<true><<<gf, 256, 0, stream>>>(
      ys_b, kD, (long)kS * kD,
      z_b,  kD, (long)kS * kD,
      out, nullptr, kD, (long)kS * kD,
      nullptr, x, (long)kS * kD,
      kS, kD, kS, /*flags=*/0);
}
